// SpikeToGraphDeepONet_11003706212819
// MI455X (gfx1250) — matrix operand order repaired and checked by reading, not compiled, not run on hardware
//
#include <hip/hip_runtime.h>
#include <hip/hip_bf16.h>

typedef __attribute__((ext_vector_type(16))) __bf16 v16bf;
typedef __attribute__((ext_vector_type(8)))  float  v8f;

struct alignas(16) U4 { unsigned int x, y, z, w; };

union Frag {
  v16bf v;
  __bf16 h[16];
  U4 q[2];
};

#define NB   512
#define TLEN 1000

// ---------------- workspace layout (bytes) ----------------
// W2f    : 64*5*64 bf16   @ 0        (40960)   [co][k][ci]
// W3f    : 64*5*64 bf16   @ 40960    (40960)   [co][k][ci]
// Wt1T   : 128*32 bf16    @ 81920    (8192)    [p][k] (k padded 9->32)
// Wt2T   : 128*128 bf16   @ 90112    (32768)   [p_out][k_in]
// Wt3T   : 128*128 bf16   @ 122880   (32768)   [p_out][k_in]
//   (Wt1T|Wt2T|Wt3T contiguous: 73728 bytes @ 81920 -> one linear LDS fill)
// rowmean: 1024*64 f32    @ 155648   (262144)
// cvec   : 2*128 f32      @ 417792   (1024)

// ============ weight prep: fp32 -> bf16, WMMA-friendly layouts ============
__global__ void __launch_bounds__(256)
prep_kernel(const float* __restrict__ W2, const float* __restrict__ W3,
            const float* __restrict__ Wt1, const float* __restrict__ Wt2,
            const float* __restrict__ Wt3,
            __bf16* __restrict__ W2f, __bf16* __restrict__ W3f,
            __bf16* __restrict__ Wt1T, __bf16* __restrict__ Wt2T,
            __bf16* __restrict__ Wt3T)
{
  int idx = blockIdx.x * 256 + threadIdx.x;
  if (idx < 20480) {                      // W2f[co][k][ci] <- W2[co][ci][k]
    int co = idx / 320, r = idx % 320, k = r >> 6, ci = r & 63;
    W2f[idx] = (__bf16)W2[(co * 64 + ci) * 5 + k];
  } else if (idx < 40960) {
    int t = idx - 20480;
    int co = t / 320, r = t % 320, k = r >> 6, ci = r & 63;
    W3f[t] = (__bf16)W3[(co * 64 + ci) * 5 + k];
  } else if (idx < 45056) {               // Wt1T[p][k], k padded to 32
    int t = idx - 40960; int p = t >> 5, k = t & 31;
    Wt1T[t] = (k < 9) ? (__bf16)Wt1[k * 128 + p] : (__bf16)0.f;
  } else if (idx < 61440) {               // Wt2T[p][k] <- Wt2[k][p]
    int t = idx - 45056; int p = t >> 7, k = t & 127;
    Wt2T[t] = (__bf16)Wt2[k * 128 + p];
  } else if (idx < 77824) {               // Wt3T[p][k] <- Wt3[k][p]
    int t = idx - 61440; int p = t >> 7, k = t & 127;
    Wt3T[t] = (__bf16)Wt3[k * 128 + p];
  }
}

// ============ branch conv stack: conv1 scalar, conv2/3 as shifted WMMA GEMMs ============
__global__ void __launch_bounds__(128)
branch_conv_kernel(const float* __restrict__ spikes,
                   const float* __restrict__ W1, const float* __restrict__ b1,
                   const float* __restrict__ b2, const float* __restrict__ b3,
                   const __bf16* __restrict__ W2f, const __bf16* __restrict__ W3f,
                   float* __restrict__ rowmean)
{
  __shared__ float xs[140];
  __shared__ float w1s[320];
  __shared__ float b1s[64];
  __shared__ __align__(16) __bf16 h1[136][72];   // [time][ci], padded stride
  __shared__ __align__(16) __bf16 h2[128][72];

  const int row   = blockIdx.x;          // b*N + n
  const int tid   = threadIdx.x;
  const int lane  = tid & 31;
  const int wv    = tid >> 5;            // 4 waves, each owns 16 out-channels
  const int mbase = wv * 16;
  const int lrow  = lane & 15;
  const int lhalf = lane >> 4;
  const float* x  = spikes + (long)row * TLEN;

  for (int i = tid; i < 320; i += 128) w1s[i] = W1[i];
  if (tid < 64) b1s[tid] = b1[tid];

  // hoist conv weight A-fragments into registers (constant across time tiles)
  Frag aw2[5][2], aw3[5][2];
#pragma unroll
  for (int s = 0; s < 5; ++s)
#pragma unroll
    for (int kh = 0; kh < 2; ++kh) {
      const int m   = mbase + lrow;
      const int ci0 = kh * 32 + lhalf * 8;     // A-layout K pattern {0-7,16-23}/{8-15,24-31}
      const __bf16* p2 = W2f + (m * 5 + s) * 64 + ci0;
      aw2[s][kh].q[0] = *(const U4*)p2;
      aw2[s][kh].q[1] = *(const U4*)(p2 + 16);
      const __bf16* p3 = W3f + (m * 5 + s) * 64 + ci0;
      aw3[s][kh].q[0] = *(const U4*)p3;
      aw3[s][kh].q[1] = *(const U4*)(p3 + 16);
    }
  float bias2[8], bias3[8];
#pragma unroll
  for (int r = 0; r < 8; ++r) {
    bias2[r] = b2[mbase + r + lhalf * 8];
    bias3[r] = b3[mbase + r + lhalf * 8];
  }

  float acc[8];
#pragma unroll
  for (int r = 0; r < 8; ++r) acc[r] = 0.f;

  for (int t0 = 0; t0 < TLEN; t0 += 112) {
    __syncthreads();
    // stage spikes segment [t0-6, t0+134) with zero padding
    for (int i = tid; i < 140; i += 128) {
      int t = t0 - 6 + i;
      xs[i] = (t >= 0 && t < TLEN) ? x[t] : 0.f;
    }
    __syncthreads();
    // conv1 (1->64), h1 local time base = t0-4
    for (int idx = tid; idx < 136 * 64; idx += 128) {
      int lt = idx >> 6, c = idx & 63;
      int tg = t0 - 4 + lt;
      float s = b1s[c];
#pragma unroll
      for (int k = 0; k < 5; ++k) s = fmaf(w1s[c * 5 + k], xs[lt + k], s);
      h1[lt][c] = (tg >= 0 && tg < TLEN) ? (__bf16)fmaxf(s, 0.f) : (__bf16)0.f;
    }
    __syncthreads();
    // conv2 via WMMA: 8 time tiles x (5 shifts x 2 K-halves), h2 base = t0-2
    for (int nt = 0; nt < 8; ++nt) {
      v8f c8 = {0.f, 0.f, 0.f, 0.f, 0.f, 0.f, 0.f, 0.f};
      const int col = nt * 16 + lrow;
#pragma unroll
      for (int s = 0; s < 5; ++s)
#pragma unroll
        for (int kh = 0; kh < 2; ++kh) {
          Frag bfr;
          const __bf16* p = &h1[col + s][kh * 32 + lhalf * 8];  // B-layout K pattern {0-7,16-23}/{8-15,24-31}
          bfr.q[0] = *(const U4*)p;
          bfr.q[1] = *(const U4*)(p + 16);
          c8 = __builtin_amdgcn_wmma_f32_16x16x32_bf16(
                   false, aw2[s][kh].v, false, bfr.v, (short)0, c8, false, false);
        }
      const int tg2 = t0 - 2 + col;
      const bool ok = (tg2 >= 0 && tg2 < TLEN);
#pragma unroll
      for (int r = 0; r < 8; ++r) {
        float v = fmaxf(c8[r] + bias2[r], 0.f);
        h2[col][mbase + r + lhalf * 8] = ok ? (__bf16)v : (__bf16)0.f;
      }
    }
    __syncthreads();
    // conv3 via WMMA + time pooling (7 tiles -> 112 outputs)
    for (int nt = 0; nt < 7; ++nt) {
      v8f c8 = {0.f, 0.f, 0.f, 0.f, 0.f, 0.f, 0.f, 0.f};
      const int col = nt * 16 + lrow;
#pragma unroll
      for (int s = 0; s < 5; ++s)
#pragma unroll
        for (int kh = 0; kh < 2; ++kh) {
          Frag bfr;
          const __bf16* p = &h2[col + s][kh * 32 + lhalf * 8];
          bfr.q[0] = *(const U4*)p;
          bfr.q[1] = *(const U4*)(p + 16);
          c8 = __builtin_amdgcn_wmma_f32_16x16x32_bf16(
                   false, aw3[s][kh].v, false, bfr.v, (short)0, c8, false, false);
        }
      if (t0 + col < TLEN) {
#pragma unroll
        for (int r = 0; r < 8; ++r) acc[r] += fmaxf(c8[r] + bias3[r], 0.f);
      }
    }
  }
  // reduce the 16 time-columns held across lanes (width-16 xor shuffle)
#pragma unroll
  for (int r = 0; r < 8; ++r) {
    float v = acc[r];
    v += __shfl_xor(v, 1, 16);
    v += __shfl_xor(v, 2, 16);
    v += __shfl_xor(v, 4, 16);
    v += __shfl_xor(v, 8, 16);
    if (lrow == 0)
      rowmean[row * 64 + mbase + r + lhalf * 8] = v * (1.0f / TLEN);
  }
}

// ============ neuron pooling + branch MLP -> c[2][128] ============
__global__ void __launch_bounds__(256)
pool_mlp_kernel(const float* __restrict__ rowmean,
                const float* __restrict__ Wm1, const float* __restrict__ bm1,
                const float* __restrict__ Wm2, const float* __restrict__ bm2,
                float* __restrict__ cvec)
{
  __shared__ float pooled[128];
  __shared__ float hbuf[128];
  const int tid = threadIdx.x;
  if (tid < 128) {
    int b = tid >> 6, c = tid & 63;
    float s = 0.f;
    for (int n = 0; n < NB; ++n) s += rowmean[(b * NB + n) * 64 + c];
    pooled[tid] = s * (1.0f / NB);
  }
  __syncthreads();
  if (tid < 128) {
    int b = tid >> 6, hc = tid & 63;
    float s = bm1[hc];
    for (int c = 0; c < 64; ++c) s = fmaf(pooled[b * 64 + c], Wm1[c * 64 + hc], s);
    hbuf[tid] = fmaxf(s, 0.f);
  }
  __syncthreads();
  {
    int b = tid >> 7, p = tid & 127;
    float s = bm2[p];
    for (int hc = 0; hc < 64; ++hc) s = fmaf(hbuf[b * 64 + hc], Wm2[hc * 128 + p], s);
    cvec[tid] = s;
  }
}

// ============ trunk MLP over edge grid + fused c.phi contraction ============
// Dynamic LDS layout (bytes):
//   Wt1s @ 0      : 128*32  bf16 ( 8192)
//   Wt2s @ 8192   : 128*128 bf16 (32768)
//   Wt3s @ 40960  : 128*128 bf16 (32768)
//   act  @ 73728  : 8*16*136 bf16 (34816)
//   total 108544
#define TRUNK_LDS_BYTES 108544

__global__ void __launch_bounds__(256)
trunk_kernel(const float* __restrict__ ei, const float* __restrict__ pos,
             const __bf16* __restrict__ WtPack,   // Wt1T|Wt2T|Wt3T contiguous (73728 B)
             const float* __restrict__ bt1, const float* __restrict__ bt2,
             const float* __restrict__ bt3,
             const float* __restrict__ cvec, float* __restrict__ out)
{
  extern __shared__ __align__(16) char smem[];
  __bf16* Wt1s = (__bf16*)(smem);
  __bf16* Wt2s = (__bf16*)(smem + 8192);
  __bf16* Wt3s = (__bf16*)(smem + 40960);
  typedef __bf16 ActRow[16][136];
  ActRow* act = (ActRow*)(smem + 73728);

  const int tid   = threadIdx.x;
  const int lane  = tid & 31;
  const int wv    = tid >> 5;
  const int lrow  = lane & 15;
  const int lhalf = lane >> 4;
  const int b     = blockIdx.z;
  const int i     = blockIdx.y;
  const int jbase = blockIdx.x * 128 + wv * 16;
  const int j     = jbase + lrow;

  // ---- cooperative stage of all trunk weights into LDS (one linear copy) ----
  {
    const U4* src = (const U4*)WtPack;
    U4* dst = (U4*)smem;
    for (int k = tid; k < 4608; k += 256) dst[k] = src[k];   // 73728 B
  }

  const float inv = 1.0f / (511.0f + 1e-8f);

  // ---- build layer-1 A fragment directly in registers (K 9 -> padded 32) ----
  Frag a1;
#pragma unroll
  for (int t = 0; t < 16; ++t) a1.h[t] = (__bf16)0.f;
  {
    float f6 = pos[b * NB + i];
    float f7 = pos[b * NB + j];
    if (lhalf == 0) {            // holds K=0..7 (and K=16..23 = pad)
      a1.h[0] = (__bf16)((float)i * inv);
      a1.h[1] = (__bf16)((float)j * inv);
      a1.h[2] = (__bf16)ei[(b * NB + i) * 2 + 0];
      a1.h[3] = (__bf16)ei[(b * NB + i) * 2 + 1];
      a1.h[4] = (__bf16)ei[(b * NB + j) * 2 + 0];
      a1.h[5] = (__bf16)ei[(b * NB + j) * 2 + 1];
      a1.h[6] = (__bf16)f6;
      a1.h[7] = (__bf16)f7;
    } else {                     // holds K=8..15 -> only K=8 (dist) nonzero
      a1.h[0] = (__bf16)fabsf(f6 - f7);
    }
  }

  __syncthreads();   // weights staged

  // ---- layer 1: [16x32] @ [32x128] ----
  for (int nt = 0; nt < 8; ++nt) {
    v8f d = {0.f, 0.f, 0.f, 0.f, 0.f, 0.f, 0.f, 0.f};
    Frag bfr;
    const __bf16* p = Wt1s + (nt * 16 + lrow) * 32 + lhalf * 8;
    bfr.q[0] = *(const U4*)p;
    bfr.q[1] = *(const U4*)(p + 16);
    d = __builtin_amdgcn_wmma_f32_16x16x32_bf16(false, a1.v, false, bfr.v,
                                                (short)0, d, false, false);
    const int pcol = nt * 16 + lrow;
    const float bias = bt1[pcol];
#pragma unroll
    for (int r = 0; r < 8; ++r)
      act[wv][r + lhalf * 8][pcol] = (__bf16)fmaxf(d[r] + bias, 0.f);
  }

  // ---- layer 2: [16x128] @ [128x128] ----
  Frag af[4];
#pragma unroll
  for (int kk = 0; kk < 4; ++kk) {
    const __bf16* p = &act[wv][lrow][kk * 32 + lhalf * 8];
    af[kk].q[0] = *(const U4*)p;
    af[kk].q[1] = *(const U4*)(p + 16);
  }
  v8f accs[8];
#pragma unroll
  for (int nt = 0; nt < 8; ++nt) {
    v8f d = {0.f, 0.f, 0.f, 0.f, 0.f, 0.f, 0.f, 0.f};
#pragma unroll
    for (int kk = 0; kk < 4; ++kk) {
      Frag bfr;
      const __bf16* p = Wt2s + (nt * 16 + lrow) * 128 + kk * 32 + lhalf * 8;
      bfr.q[0] = *(const U4*)p;
      bfr.q[1] = *(const U4*)(p + 16);
      d = __builtin_amdgcn_wmma_f32_16x16x32_bf16(false, af[kk].v, false, bfr.v,
                                                  (short)0, d, false, false);
    }
    accs[nt] = d;
  }
#pragma unroll
  for (int nt = 0; nt < 8; ++nt) {
    const int pcol = nt * 16 + lrow;
    const float bias = bt2[pcol];
#pragma unroll
    for (int r = 0; r < 8; ++r)
      act[wv][r + lhalf * 8][pcol] = (__bf16)fmaxf(accs[nt][r] + bias, 0.f);
  }

  // ---- layer 3 + fused logits = sum_p c[b,p] * phi[row,p] ----
#pragma unroll
  for (int kk = 0; kk < 4; ++kk) {
    const __bf16* p = &act[wv][lrow][kk * 32 + lhalf * 8];
    af[kk].q[0] = *(const U4*)p;
    af[kk].q[1] = *(const U4*)(p + 16);
  }
  float part[8];
#pragma unroll
  for (int r = 0; r < 8; ++r) part[r] = 0.f;
  for (int nt = 0; nt < 8; ++nt) {
    v8f d = {0.f, 0.f, 0.f, 0.f, 0.f, 0.f, 0.f, 0.f};
#pragma unroll
    for (int kk = 0; kk < 4; ++kk) {
      Frag bfr;
      const __bf16* p = Wt3s + (nt * 16 + lrow) * 128 + kk * 32 + lhalf * 8;
      bfr.q[0] = *(const U4*)p;
      bfr.q[1] = *(const U4*)(p + 16);
      d = __builtin_amdgcn_wmma_f32_16x16x32_bf16(false, af[kk].v, false, bfr.v,
                                                  (short)0, d, false, false);
    }
    const int pcol = nt * 16 + lrow;
    const float bias = bt3[pcol];
    const float cw = cvec[b * 128 + pcol];
#pragma unroll
    for (int r = 0; r < 8; ++r) part[r] += cw * (d[r] + bias);
  }
#pragma unroll
  for (int r = 0; r < 8; ++r) {
    float v = part[r];
    v += __shfl_xor(v, 1, 16);
    v += __shfl_xor(v, 2, 16);
    v += __shfl_xor(v, 4, 16);
    v += __shfl_xor(v, 8, 16);
    if (lrow == 0)
      out[((long)(b * NB + i)) * NB + jbase + r + lhalf * 8] = v;
  }
}

extern "C" void kernel_launch(void* const* d_in, const int* in_sizes, int n_in,
                              void* d_out, int out_size, void* d_ws, size_t ws_size,
                              hipStream_t stream) {
  (void)in_sizes; (void)n_in; (void)out_size; (void)ws_size;
  const float* spikes = (const float*)d_in[0];
  const float* ei     = (const float*)d_in[1];
  const float* pos    = (const float*)d_in[2];
  const float* W1  = (const float*)d_in[3];
  const float* b1  = (const float*)d_in[4];
  const float* W2  = (const float*)d_in[5];
  const float* b2  = (const float*)d_in[6];
  const float* W3  = (const float*)d_in[7];
  const float* b3  = (const float*)d_in[8];
  const float* Wm1 = (const float*)d_in[9];
  const float* bm1 = (const float*)d_in[10];
  const float* Wm2 = (const float*)d_in[11];
  const float* bm2 = (const float*)d_in[12];
  const float* Wt1 = (const float*)d_in[13];
  const float* bt1 = (const float*)d_in[14];
  const float* Wt2 = (const float*)d_in[15];
  const float* bt2 = (const float*)d_in[16];
  const float* Wt3 = (const float*)d_in[17];
  const float* bt3 = (const float*)d_in[18];

  char* ws = (char*)d_ws;
  __bf16* W2f  = (__bf16*)(ws + 0);
  __bf16* W3f  = (__bf16*)(ws + 40960);
  __bf16* Wt1T = (__bf16*)(ws + 81920);   // start of contiguous Wt1T|Wt2T|Wt3T pack
  __bf16* Wt2T = (__bf16*)(ws + 90112);
  __bf16* Wt3T = (__bf16*)(ws + 122880);
  float*  rowmean = (float*)(ws + 155648);
  float*  cvec    = (float*)(ws + 417792);

  prep_kernel<<<(77824 + 255) / 256, 256, 0, stream>>>(
      W2, W3, Wt1, Wt2, Wt3, W2f, W3f, Wt1T, Wt2T, Wt3T);
  branch_conv_kernel<<<1024, 128, 0, stream>>>(
      spikes, W1, b1, b2, b3, W2f, W3f, rowmean);
  pool_mlp_kernel<<<1, 256, 0, stream>>>(
      rowmean, Wm1, bm1, Wm2, bm2, cvec);
  trunk_kernel<<<dim3(4, 512, 2), 256, TRUNK_LDS_BYTES, stream>>>(
      ei, pos, Wt1T, bt1, bt2, bt3, cvec, (float*)d_out);
}
